// Sinkhorn_11982958756719
// MI455X (gfx1250) — compile-verified, matrix operand order checked by reading, not run on hardware
//
#include <hip/hip_runtime.h>
#include <cmath>

#define EPSV 1e-8f
#define SEG  8

typedef __attribute__((ext_vector_type(8)))  __bf16 bf16x8;
typedef __attribute__((ext_vector_type(16))) __bf16 v16bf;
typedef __attribute__((ext_vector_type(8)))  float  v8f;
typedef __attribute__((ext_vector_type(4)))  float  f32x4;

// ---------------------------------------------------------------------------
// Build K = exp(s)*m in bf16 (row major) AND K^T in bf16, via LDS-tiled
// transpose so both stores are coalesced. One-time pass.
// ---------------------------------------------------------------------------
__global__ void k_build(const float* __restrict__ s, const float* __restrict__ m,
                        __bf16* __restrict__ K, __bf16* __restrict__ KT, int N) {
    __shared__ __bf16 tile[32][33];
    const int tx = threadIdx.x, ty = threadIdx.y;
    const int bx = blockIdx.x * 32, by = blockIdx.y * 32;
#pragma unroll
    for (int r = ty; r < 32; r += 8) {
        size_t idx = (size_t)(by + r) * N + (bx + tx);
        float kv = __expf(s[idx]) * m[idx];
        __bf16 kb = (__bf16)kv;
        K[idx] = kb;
        tile[r][tx] = kb;
    }
    __syncthreads();
#pragma unroll
    for (int r = ty; r < 32; r += 8) {
        size_t idx = (size_t)(bx + r) * N + (by + tx);
        KT[idx] = tile[tx][r];
    }
}

// ---------------------------------------------------------------------------
// v0 = 1/N in both f32 and bf16.
// ---------------------------------------------------------------------------
__global__ void k_init(float* __restrict__ vf, __bf16* __restrict__ vb, int N) {
    int i = blockIdx.x * blockDim.x + threadIdx.x;
    if (i < N) {
        float x = 1.0f / (float)N;
        vf[i] = x;
        vb[i] = (__bf16)x;
    }
}

// ---------------------------------------------------------------------------
// WMMA matvec partial: y_part[seg, row] = sum_{k in segment} M[row,k] * x[k].
// One wave owns a 16-row tile and one K-segment. A = contiguous bf16 row tile,
// B = x broadcast into all 16 columns, accumulated in f32 with
// v_wmma_f32_16x16x32_bf16. Every column of D holds the result; lanes 0/16
// (column N=0, rows M=0..7 / 8..15) write it out.
// ---------------------------------------------------------------------------
__global__ void __launch_bounds__(256)
k_matvec(const __bf16* __restrict__ M, const __bf16* __restrict__ xb,
         float* __restrict__ part, int N) {
    const int lane = threadIdx.x & 31;
    const int wave = threadIdx.x >> 5;
    const int tile = (blockIdx.x * 8 + wave) * 16;          // row-tile base
    const int seg  = blockIdx.y;
    const int segLen = N / SEG;
    const int kbeg = seg * segLen;
    const int kend = kbeg + segLen;

    const int row  = tile + (lane & 15);
    const int koff = (lane < 16) ? 0 : 8;                   // 16-bit A layout
    const __bf16* rowp = M + (size_t)row * N;

    v8f c = {0.f, 0.f, 0.f, 0.f, 0.f, 0.f, 0.f, 0.f};

    for (int k0 = kbeg; k0 < kend; k0 += 32) {
        __builtin_prefetch(rowp + k0 + 512, 0, 1);          // global_prefetch_b8
        bf16x8 alo = *(const bf16x8*)(rowp + k0 + koff);
        bf16x8 ahi = *(const bf16x8*)(rowp + k0 + 16 + koff);
        bf16x8 blo = *(const bf16x8*)(xb + k0 + koff);
        bf16x8 bhi = *(const bf16x8*)(xb + k0 + 16 + koff);
        v16bf a, b;
#pragma unroll
        for (int t = 0; t < 8; ++t) {
            a[t] = alo[t];  a[8 + t] = ahi[t];
            b[t] = blo[t];  b[8 + t] = bhi[t];              // broadcast x into B
        }
        c = __builtin_amdgcn_wmma_f32_16x16x32_bf16(
                /*neg_a=*/false, a, /*neg_b=*/false, b,
                /*c_mod=*/(short)0, c, /*reuse_a=*/false, /*reuse_b=*/false);
    }

    if ((lane & 15) == 0) {
        int rbase = tile + ((lane >> 4) << 3);              // lane0->M=0..7, lane16->M=8..15
        float* p = part + (size_t)seg * N + rbase;
#pragma unroll
        for (int r = 0; r < 8; ++r) p[r] = c[r];
    }
}

// ---------------------------------------------------------------------------
// y[row] = 1/(sum_seg part + eps), stored as f32 (final pass) and bf16 (next
// matvec's B operand).
// ---------------------------------------------------------------------------
__global__ void k_reduce(const float* __restrict__ part, float* __restrict__ yf,
                         __bf16* __restrict__ yb, int N) {
    int r = blockIdx.x * blockDim.x + threadIdx.x;
    if (r < N) {
        float sum = 0.f;
#pragma unroll
        for (int s = 0; s < SEG; ++s) sum += part[(size_t)s * N + r];
        float val = 1.0f / (sum + EPSV);
        yf[r] = val;
        yb[r] = (__bf16)val;
    }
}

// ---------------------------------------------------------------------------
// out = u[i] * exp(s)*m * v[j], fp32, recomputed from inputs for full
// precision of K in the output. float4 vectorized.
// ---------------------------------------------------------------------------
__global__ void k_final(const float* __restrict__ s, const float* __restrict__ m,
                        const float* __restrict__ u, const float* __restrict__ v,
                        float* __restrict__ out, int N) {
    size_t t = (size_t)blockIdx.x * blockDim.x + threadIdx.x;
    size_t base = t * 4;
    size_t NN = (size_t)N * N;
    if (base >= NN) return;
    int i = (int)(base / (size_t)N);
    int j = (int)(base - (size_t)i * N);
    f32x4 s4 = *(const f32x4*)(s + base);
    f32x4 m4 = *(const f32x4*)(m + base);
    f32x4 v4 = *(const f32x4*)(v + j);
    float ui = u[i];
    f32x4 o;
#pragma unroll
    for (int q = 0; q < 4; ++q) o[q] = ui * __expf(s4[q]) * m4[q] * v4[q];
    *(f32x4*)(out + base) = o;
}

// ---------------------------------------------------------------------------
extern "C" void kernel_launch(void* const* d_in, const int* in_sizes, int n_in,
                              void* d_out, int out_size, void* d_ws, size_t ws_size,
                              hipStream_t stream) {
    const float* s = (const float*)d_in[0];
    long long half = (long long)in_sizes[0] / 2;
    int N = (int)(std::sqrt((double)half) + 0.5);           // 8192
    size_t NN = (size_t)N * N;
    const float* m = s + NN;

    char* ws = (char*)d_ws;
    __bf16* K   = (__bf16*)ws;                              // 2*NN bytes
    __bf16* KT  = (__bf16*)(ws + 2 * NN);                   // 2*NN bytes
    float*  uf  = (float*)(ws + 4 * NN);                    // N floats
    float*  vf  = uf + N;                                   // N floats
    __bf16* ub  = (__bf16*)(vf + N);                        // N bf16
    __bf16* vb  = ub + N;                                   // N bf16
    float*  part = (float*)((char*)(vb + N));               // SEG*N floats
    float*  out = (float*)d_out;

    // 1) Build bf16 K and K^T (both L2-friendly, 128 MB each).
    dim3 bblk(32, 8), bgrd(N / 32, N / 32);
    k_build<<<bgrd, bblk, 0, stream>>>(s, m, K, KT, N);

    // 2) v = 1/N.
    k_init<<<(N + 255) / 256, 256, 0, stream>>>(vf, vb, N);

    // 3) 20 Sinkhorn iterations: u = 1/(K v + eps); v = 1/(K^T u + eps).
    dim3 mvgrd(N / (16 * 8), SEG);                          // 8 waves/block, SEG k-segments
    int rblk = (N + 255) / 256;
    for (int it = 0; it < 20; ++it) {
        k_matvec<<<mvgrd, 256, 0, stream>>>(K,  vb, part, N);
        k_reduce<<<rblk, 256, 0, stream>>>(part, uf, ub, N);
        k_matvec<<<mvgrd, 256, 0, stream>>>(KT, ub, part, N);
        k_reduce<<<rblk, 256, 0, stream>>>(part, vf, vb, N);
    }

    // 4) out = u K v (K recomputed in fp32 from inputs).
    size_t nthreads = NN / 4;
    k_final<<<(nthreads + 255) / 256, 256, 0, stream>>>(s, m, uf, vf, out, N);
}
